// NeuromorphicLM_33148557590860
// MI455X (gfx1250) — matrix operand order repaired and checked by reading, not compiled, not run on hardware
//
#include <hip/hip_runtime.h>
#include <hip/hip_bf16.h>
#include <cmath>

typedef __attribute__((ext_vector_type(16))) __bf16 v16bf;
typedef __attribute__((ext_vector_type(8)))  __bf16 v8bf;
typedef __attribute__((ext_vector_type(4)))  __bf16 v4bf;
typedef __attribute__((ext_vector_type(8)))  float  v8f;
typedef __attribute__((ext_vector_type(4)))  float  v4f;

namespace {

constexpr int kBS = 4, kSEQ = 512, kD = 1024, kDI = 2048;
constexpr int kC = 16, kDC = 64;
constexpr int kV = 32000, kB = 4, kDPM = 256, kM = 2048;
constexpr int kTRAIL = 2, kTOPK = 8;
constexpr int kTOK = kBS * kSEQ; // 2048

__device__ __forceinline__ v4bf cvt4(v4f f) {
  v4bf h;
  h[0] = (__bf16)f[0]; h[1] = (__bf16)f[1];
  h[2] = (__bf16)f[2]; h[3] = (__bf16)f[3];
  return h;
}

// ------------------------------------------------------------------
// bf16 WMMA GEMM:  C[M,N] (+)= alpha * A[M,K] x op(B)
// op(B) = B[K,N] (transB=0) or B[N,K] row-major (transB=1).
// Block tile 128x64, 8 waves; wave wv owns M-subtile wv and all 4 N-subtiles
// (A fragment reused 4x). Register-double-buffered global->LDS staging with
// float4 loads and packed bf16 stores. Batched via ((z/div)%mod)*stride.
// ------------------------------------------------------------------
__global__ void gemm_wmma_bf16(const float* __restrict__ A, const float* __restrict__ B,
                               float* __restrict__ C,
                               int K, int lda, int ldb, int ldc,
                               int transB, int accum, float alpha,
                               int adiv, int amod, long long astr,
                               int bdiv, int bmod, long long bstr,
                               int cdiv, int cmod, long long cstr)
{
  __shared__ alignas(64) __bf16 As[128][32];   // [m][k]
  __shared__ alignas(64) __bf16 Bs[64][32];    // [n][k]
  const int z = blockIdx.z;
  A += (long long)((z / adiv) % amod) * astr;
  B += (long long)((z / bdiv) % bmod) * bstr;
  C += (long long)((z / cdiv) % cmod) * cstr;
  const int m0 = blockIdx.y * 128, n0 = blockIdx.x * 64;
  const int tid = threadIdx.x, lane = tid & 31, wv = tid >> 5;
  const int l15 = lane & 15;
  const int kbA = (lane < 16) ? 0 : 16;   // A frag K-base (contiguous, matches B)
  const int kbB = (lane < 16) ? 0 : 16;   // B frag K-base (contiguous, matches A)

  // staging coordinates (fixed per thread)
  const int am = tid >> 3, ak = (tid & 7) * 4;     // A: rows am+32s, quad ak
  const int bknt = tid >> 4, bnq = (tid & 15) * 4; // B no-trans: k rows, n quad

  // per-thread streaming pointers
  const float* pA  = A + (long long)(m0 + am) * lda + ak;            // += 32
  const float* pBt = B + (long long)(n0 + am) * ldb + ak;            // += 32 (transB)
  const float* pBn = B + (long long)bknt * ldb + (n0 + bnq);         // += 32*ldb
  const long long aRow32 = 32LL * lda;
  const long long bRow16 = 16LL * ldb;

  v4f ra[4], rb[2];
  auto load_tile = [&]() {
    ra[0] = *(const v4f*)(pA);
    ra[1] = *(const v4f*)(pA + aRow32);
    ra[2] = *(const v4f*)(pA + 2 * aRow32);
    ra[3] = *(const v4f*)(pA + 3 * aRow32);
    if (transB) {
      rb[0] = *(const v4f*)(pBt);
      rb[1] = *(const v4f*)(pBt + aRow32);      // rows n0+am and n0+am+32
      pBt += 32;
    } else {
      rb[0] = *(const v4f*)(pBn);
      rb[1] = *(const v4f*)(pBn + bRow16);      // k rows bknt and bknt+16
      pBn += 2 * bRow16;
    }
    pA += 32;
  };
  auto stage_tile = [&]() {
    *(v4bf*)&As[am][ak]      = cvt4(ra[0]);
    *(v4bf*)&As[am + 32][ak] = cvt4(ra[1]);
    *(v4bf*)&As[am + 64][ak] = cvt4(ra[2]);
    *(v4bf*)&As[am + 96][ak] = cvt4(ra[3]);
    if (transB) {
      *(v4bf*)&Bs[am][ak]      = cvt4(rb[0]);
      *(v4bf*)&Bs[am + 32][ak] = cvt4(rb[1]);
    } else {
      v4bf h0 = cvt4(rb[0]), h1 = cvt4(rb[1]);
      Bs[bnq + 0][bknt] = h0[0]; Bs[bnq + 1][bknt] = h0[1];
      Bs[bnq + 2][bknt] = h0[2]; Bs[bnq + 3][bknt] = h0[3];
      Bs[bnq + 0][bknt + 16] = h1[0]; Bs[bnq + 1][bknt + 16] = h1[1];
      Bs[bnq + 2][bknt + 16] = h1[2]; Bs[bnq + 3][bknt + 16] = h1[3];
    }
  };

  v8f c0 = {}, c1 = {}, c2 = {}, c3 = {};
  load_tile();
  for (int k0 = 0; k0 < K; k0 += 32) {
    __syncthreads();            // LDS consumed by previous iteration
    stage_tile();
    __syncthreads();
    if (k0 + 32 < K) {
      load_tile();              // next tile in flight during WMMA
      if (k0 + 64 < K) __builtin_prefetch(pA + 32, 0, 3);
    }
    // A fragment: lane<16 holds K 0..15, lane>=16 K 16..31
    const int mrow = wv * 16 + l15;
    v8bf alo = *(const v8bf*)&As[mrow][kbA];
    v8bf ahi = *(const v8bf*)&As[mrow][kbA + 8];
    v16bf af = __builtin_shufflevector(alo, ahi,
                0,1,2,3,4,5,6,7,8,9,10,11,12,13,14,15);
    // B fragments: lane<16 holds K 0..15 of column N=lane, lane>=16 K 16..31
    v16bf b0 = *(const v16bf*)&Bs[ 0 + l15][kbB];
    v16bf b1 = *(const v16bf*)&Bs[16 + l15][kbB];
    v16bf b2 = *(const v16bf*)&Bs[32 + l15][kbB];
    v16bf b3 = *(const v16bf*)&Bs[48 + l15][kbB];
    c0 = __builtin_amdgcn_wmma_f32_16x16x32_bf16(false, af, false, b0, (short)0, c0, false, false);
    c1 = __builtin_amdgcn_wmma_f32_16x16x32_bf16(false, af, false, b1, (short)0, c1, false, false);
    c2 = __builtin_amdgcn_wmma_f32_16x16x32_bf16(false, af, false, b2, (short)0, c2, false, false);
    c3 = __builtin_amdgcn_wmma_f32_16x16x32_bf16(false, af, false, b3, (short)0, c3, false, false);
  }
  // C/D layout: c[v] @ lane L -> row = v + (L<16?0:8), col = L%16
  const int rb0 = (lane < 16) ? 0 : 8;
  #pragma unroll
  for (int v = 0; v < 8; v++) {
    const int row = m0 + wv * 16 + rb0 + v;
    long long i0 = (long long)row * ldc + (n0 + l15);
    float r0 = alpha * c0[v], r1 = alpha * c1[v];
    float r2 = alpha * c2[v], r3 = alpha * c3[v];
    if (accum) {
      C[i0] += r0; C[i0 + 16] += r1; C[i0 + 32] += r2; C[i0 + 48] += r3;
    } else {
      C[i0] = r0; C[i0 + 16] = r1; C[i0 + 32] = r2; C[i0 + 48] = r3;
    }
  }
}

// ---------------- embedding gather + positional add ----------------
__global__ void embed_kernel(const int* __restrict__ ids, const float* __restrict__ emb,
                             const float* __restrict__ pos, float* __restrict__ X,
                             float* __restrict__ H)
{
  const int tok = blockIdx.x, n = tok % kSEQ;
  const int id = ids[tok];
  const int d = threadIdx.x * 4;   // 256 threads x vec4 == 1024
  v4f e = *(const v4f*)&emb[(long long)id * kD + d];
  v4f p = *(const v4f*)&pos[(long long)n * kD + d];
  v4f v = e + p;
  *(v4f*)&X[(long long)tok * kD + d] = v;
  *(v4f*)&H[(long long)tok * kD + d] = v;
}

// ---------------- row LayerNorm over D=1024 (1 vec4 per thread) ----------------
__global__ void ln_rows(const float* __restrict__ X, const float* __restrict__ g,
                        const float* __restrict__ b, float* __restrict__ Y)
{
  const int row = blockIdx.x, tid = threadIdx.x;
  const int d = tid * 4;
  __shared__ float red[256];
  v4f xv = *(const v4f*)&X[(long long)row * kD + d];
  red[tid] = xv[0] + xv[1] + xv[2] + xv[3];
  __syncthreads();
  for (int o = 128; o > 0; o >>= 1) { if (tid < o) red[tid] += red[tid + o]; __syncthreads(); }
  const float mu = red[0] * (1.f / kD);
  __syncthreads();
  v4f dv = xv - mu;
  red[tid] = dv[0]*dv[0] + dv[1]*dv[1] + dv[2]*dv[2] + dv[3]*dv[3];
  __syncthreads();
  for (int o = 128; o > 0; o >>= 1) { if (tid < o) red[tid] += red[tid + o]; __syncthreads(); }
  const float rstd = rsqrtf(red[0] * (1.f / kD) + 1e-5f);
  v4f gv = *(const v4f*)&g[d];
  v4f bv = *(const v4f*)&b[d];
  *(v4f*)&Y[(long long)row * kD + d] = dv * rstd * gv + bv;
}

// ------- gated linear recurrence; u <- h_t * sigmoid(gpre), in place -------
__global__ void scan_kernel(float* __restrict__ u, const float* __restrict__ gpre,
                            const float* __restrict__ Alog)
{
  const int idx = blockIdx.x * blockDim.x + threadIdx.x;  // kBS*kDI threads
  const int b = idx / kDI, ch = idx % kDI;
  const float a = 1.f / (1.f + __expf(-Alog[ch]));
  float h = 0.f;
  const long long base = (long long)b * kSEQ * kDI + ch;
  float uv = u[base], gv = gpre[base];
  for (int t = 0; t < kSEQ; t++) {
    const long long o = base + (long long)t * kDI;
    float un = 0.f, gn = 0.f;
    if (t + 1 < kSEQ) { un = u[o + kDI]; gn = gpre[o + kDI]; }  // prefetch next
    h = a * h + (1.f - a) * uv;
    const float gate = 1.f / (1.f + __expf(-gv));
    u[o] = h * gate;
    uv = un; gv = gn;
  }
}

// -------- PCM per-column predict/surprise/gain + seed generation --------
__global__ void pcm_kernel(float* __restrict__ H, const float* __restrict__ X,
                           const float* __restrict__ Wp, const float* __restrict__ gs,
                           const float* __restrict__ Wsw, const float* __restrict__ bsw,
                           float* __restrict__ seed, float* __restrict__ aux, float auxscale)
{
  const int c = blockIdx.x & (kC - 1);
  const int tok = blockIdx.x >> 4;
  const int e = threadIdx.x;                    // 64 threads
  __shared__ float hcol[kDC], hupd[kDC], red[kDC];
  const long long base = (long long)tok * kD + c * kDC;
  hcol[e] = H[base + e];
  __syncthreads();
  const float* wp = Wp + (long long)c * kDC * kDC;
  float z = 0.f;
  for (int d = 0; d < kDC; d++) z += hcol[d] * wp[d * kDC + e];
  const float xv = X[base + e];
  const float hn = hcol[e] * (1.f + gs[c] * tanhf(xv - z));
  H[base + e] = hn;
  hupd[e] = hn;
  const float dd = z - xv;
  red[e] = dd * dd;
  __syncthreads();
  for (int o = 32; o > 0; o >>= 1) { if (e < o) red[e] += red[e + o]; __syncthreads(); }
  if (e == 0) atomicAdd(aux, red[0] * auxscale);
  const float* wsw = Wsw + (long long)c * kDC * (2 * kDC);
  float sv = bsw[c * 2 * kDC + e];
  for (int d = 0; d < kDC; d++) sv += hupd[d] * wsw[d * (2 * kDC) + e];
  seed[base + e] = sv;
}

__global__ void gelu_kernel(float* __restrict__ p, int n)
{
  const int i = (blockIdx.x * blockDim.x + threadIdx.x) * 4;
  if (i < n) {
    v4f x = *(const v4f*)&p[i];
    v4f y;
    #pragma unroll
    for (int j = 0; j < 4; j++) {
      float t = 0.7978845608028654f * (x[j] + 0.044715f * x[j] * x[j] * x[j]);
      y[j] = 0.5f * x[j] * (1.f + tanhf(t));
    }
    *(v4f*)&p[i] = y;
  }
}

// -------- EM: strength-scale, top-8 via iterative argmax, softmax, gather --------
__global__ void em_read_kernel(const float* __restrict__ scores, const float* __restrict__ strength,
                               const float* __restrict__ vals, float* __restrict__ read_t,
                               float* __restrict__ read_acc, int first)
{
  const int gid = blockIdx.x;     // (b*4+k)*512 + n
  const int bk = gid >> 9;
  const int tid = threadIdx.x;
  __shared__ float sc[kM];
  __shared__ float rmax[256];
  __shared__ int   rmaxi[256];
  __shared__ float wsel[kTOPK];
  __shared__ int   isel[kTOPK];
  const float* sg = scores + (long long)gid * kM;
  const float* st = strength + (long long)bk * kM;
  {
    const int s = tid * 4;                      // 2048 = 256 * 8, two vec4
    v4f a0 = *(const v4f*)&sg[s];
    v4f b0 = *(const v4f*)&st[s];
    v4f a1 = *(const v4f*)&sg[s + 1024];
    v4f b1 = *(const v4f*)&st[s + 1024];
    *(v4f*)&sc[s] = a0 * b0;
    *(v4f*)&sc[s + 1024] = a1 * b1;
  }
  __syncthreads();
  for (int j = 0; j < kTOPK; j++) {
    float m = -3.4e38f; int mi = 0;
    for (int s = tid; s < kM; s += 256) { float v = sc[s]; if (v > m) { m = v; mi = s; } }
    rmax[tid] = m; rmaxi[tid] = mi;
    __syncthreads();
    for (int o = 128; o > 0; o >>= 1) {
      if (tid < o && rmax[tid + o] > rmax[tid]) { rmax[tid] = rmax[tid + o]; rmaxi[tid] = rmaxi[tid + o]; }
      __syncthreads();
    }
    if (tid == 0) { wsel[j] = rmax[0]; isel[j] = rmaxi[0]; sc[rmaxi[0]] = -3.4e38f; }
    __syncthreads();
  }
  if (tid == 0) {
    float mx = wsel[0];
    for (int j = 1; j < kTOPK; j++) mx = fmaxf(mx, wsel[j]);
    float ss = 0.f;
    for (int j = 0; j < kTOPK; j++) { wsel[j] = __expf(wsel[j] - mx); ss += wsel[j]; }
    float inv = 1.f / ss;
    for (int j = 0; j < kTOPK; j++) wsel[j] *= inv;
  }
  __syncthreads();
  const float* vb = vals + (long long)bk * kM * kD;
  const long long ob = (long long)gid * kD;
  const int d = tid * 4;                        // 1024 = 256 * 4
  v4f acc = {};
  #pragma unroll
  for (int j = 0; j < kTOPK; j++) {
    v4f vv = *(const v4f*)&vb[(long long)isel[j] * kD + d];
    acc += wsel[j] * vv;
  }
  *(v4f*)&read_t[ob + d] = acc;
  if (first) *(v4f*)&read_acc[ob + d] = acc;
  else { v4f prev = *(const v4f*)&read_acc[ob + d]; *(v4f*)&read_acc[ob + d] = prev + acc; }
}

// -------- H += pm_read + sum_k read_acc --------
__global__ void combine_kernel(float* __restrict__ H, const float* __restrict__ pmr,
                               const float* __restrict__ ra)
{
  const int tok = blockIdx.x;
  const int b = tok / kSEQ, n = tok % kSEQ;
  const int d = threadIdx.x * 4;
  v4f s = *(const v4f*)&H[(long long)tok * kD + d];
  s += *(const v4f*)&pmr[(long long)tok * kD + d];
  #pragma unroll
  for (int k = 0; k < kB; k++)
    s += *(const v4f*)&ra[((long long)(b * kB + k) * kSEQ + n) * kD + d];
  *(v4f*)&H[(long long)tok * kD + d] = s;
}

__global__ void zero1_kernel(float* p) { if (threadIdx.x == 0 && blockIdx.x == 0) *p = 0.f; }

// ---------------- host-side helpers ----------------
inline void launch_gemm(hipStream_t s, const float* A, const float* B, float* C,
                        int M, int Nn, int K, int lda, int ldb, int ldc,
                        int transB, int accum, float alpha,
                        int adiv, int amod, long long astr,
                        int bdiv, int bmod, long long bstr,
                        int cdiv, int cmod, long long cstr, int batch)
{
  dim3 grid(Nn / 64, M / 128, batch);
  gemm_wmma_bf16<<<grid, dim3(256), 0, s>>>(A, B, C, K, lda, ldb, ldc, transB, accum, alpha,
                                            adiv, amod, astr, bdiv, bmod, bstr, cdiv, cmod, cstr);
}

inline void scan_layer(hipStream_t s, float* H, float* xn, float* u, float* g,
                       const float* lng, const float* lnb, const float* Wi,
                       const float* Wg, const float* Al, const float* Wo)
{
  ln_rows<<<kTOK, 256, 0, s>>>(H, lng, lnb, xn);
  launch_gemm(s, xn, Wi, u, kTOK, kDI, kD, kD, kDI, kDI, 0, 0, 1.f,
              1, 1, 0, 1, 1, 0, 1, 1, 0, 1);
  launch_gemm(s, xn, Wg, g, kTOK, kDI, kD, kD, kDI, kDI, 0, 0, 1.f,
              1, 1, 0, 1, 1, 0, 1, 1, 0, 1);
  scan_kernel<<<(kBS * kDI) / 256, 256, 0, s>>>(u, g, Al);
  launch_gemm(s, u, Wo, H, kTOK, kD, kDI, kDI, kD, kD, 0, 1, 1.f,
              1, 1, 0, 1, 1, 0, 1, 1, 0, 1);   // residual via accumulate
}

} // namespace

extern "C" void kernel_launch(void* const* d_in, const int* in_sizes, int n_in,
                              void* d_out, int out_size, void* d_ws, size_t ws_size,
                              hipStream_t stream) {
  (void)in_sizes; (void)n_in; (void)ws_size;
  const int*   ids     = (const int*)  d_in[0];
  const float* embed   = (const float*)d_in[1];
  const float* pos     = (const float*)d_in[2];
  const float* ln_g    = (const float*)d_in[3];
  const float* ln_b    = (const float*)d_in[4];
  const float* W_in    = (const float*)d_in[5];
  const float* W_g     = (const float*)d_in[6];
  const float* A_log   = (const float*)d_in[7];
  const float* W_out   = (const float*)d_in[8];
  const float* W_pred  = (const float*)d_in[9];
  const float* g_scale = (const float*)d_in[10];
  const float* W_sw    = (const float*)d_in[11];
  const float* b_sw    = (const float*)d_in[12];
  const float* pm_W1   = (const float*)d_in[13];
  const float* pm_W2   = (const float*)d_in[14];
  const float* em_keys = (const float*)d_in[15];
  const float* em_vals = (const float*)d_in[16];
  const float* em_str  = (const float*)d_in[17];
  const float* em_Wq   = (const float*)d_in[18];
  const float* lnf_g   = (const float*)d_in[19];
  const float* lnf_b   = (const float*)d_in[20];

  float* logits = (float*)d_out;
  float* aux    = (float*)d_out + (out_size - 1);

  // workspace layout (floats)
  float* ws = (float*)d_ws;
  size_t o = 0;
  float* xbuf = ws + o; o += (size_t)kTOK * kD;        // x (kept for PCM)
  float* Hb   = ws + o; o += (size_t)kTOK * kD;        // running hidden state
  float* xn   = ws + o; o += (size_t)kTOK * kD;        // layernorm output
  float* ub   = ws + o; o += (size_t)kTOK * kDI;       // u / h*gate
  float* gb   = ws + o; o += (size_t)kTOK * kDI;       // pre-sigmoid gate
  float* seed = ws + o; o += (size_t)kTOK * kD;
  float* pmp  = ws + o; o += (size_t)kTOK * kDPM;      // pm pre-activation
  float* pmr  = ws + o; o += (size_t)kTOK * kD;        // pm read
  float* qb   = ws + o; o += (size_t)kBS * kB * kSEQ * 64;
  float* scb  = ws + o; o += (size_t)kBS * kB * kSEQ * kM;
  float* rt   = ws + o; o += (size_t)kBS * kB * kSEQ * kD;
  float* ra   = ws + o; o += (size_t)kBS * kB * kSEQ * kD;

  // 1) embedding + positional
  embed_kernel<<<kTOK, 256, 0, stream>>>(ids, embed, pos, xbuf, Hb);

  // 2) first 4 scan layers
  for (int l = 0; l < 4; l++)
    scan_layer(stream, Hb, xn, ub, gb,
               ln_g + (size_t)l * kD, ln_b + (size_t)l * kD,
               W_in + (size_t)l * kD * kDI, W_g + (size_t)l * kD * kDI,
               A_log + (size_t)l * kDI, W_out + (size_t)l * kDI * kD);

  // 3) PCM predict/surprise/gain + seed; aux loss
  zero1_kernel<<<1, 64, 0, stream>>>(aux);
  const float auxscale = 0.1f / (float)((size_t)kTOK * kD);
  pcm_kernel<<<kTOK * kC, kDC, 0, stream>>>(Hb, xbuf, W_pred, g_scale, W_sw, b_sw,
                                            seed, aux, auxscale);

  // 4) PM fast-weight read: gelu(H @ pm_W1) @ pm_W2, per batch
  launch_gemm(stream, Hb, pm_W1, pmp, kSEQ, kDPM, kD, kD, kDPM, kDPM, 0, 0, 1.f,
              1, 4, (long long)kSEQ * kD, 1, 4, (long long)kD * kDPM,
              1, 4, (long long)kSEQ * kDPM, 4);
  gelu_kernel<<<(kTOK * kDPM) / 1024, 256, 0, stream>>>(pmp, kTOK * kDPM);
  launch_gemm(stream, pmp, pm_W2, pmr, kSEQ, kD, kDPM, kDPM, kD, kD, 0, 0, 1.f,
              1, 4, (long long)kSEQ * kDPM, 1, 4, (long long)kDPM * kD,
              1, 4, (long long)kSEQ * kD, 4);

  // 5) EM trail read. q = seed @ em_Wq  (z = b*4+k)
  launch_gemm(stream, seed, em_Wq, qb, kSEQ, 64, kD, kD, 64, 64, 0, 0, 1.f,
              4, 4, (long long)kSEQ * kD, 1, 4, (long long)kD * 64,
              1, 16, (long long)kSEQ * 64, 16);
  for (int t = 0; t < kTRAIL; t++) {
    // scores = q x keys^T  (keys row-major [slots, D_mem])
    launch_gemm(stream, qb, em_keys, scb, kSEQ, kM, 64, 64, 64, kM, 1, 0, 1.f,
                1, 16, (long long)kSEQ * 64, 1, 16, (long long)kM * 64,
                1, 16, (long long)kSEQ * kM, 16);
    em_read_kernel<<<kBS * kB * kSEQ, 256, 0, stream>>>(scb, em_str, em_vals, rt, ra,
                                                        (t == 0) ? 1 : 0);
    if (t + 1 < kTRAIL) {
      // q += read_t @ em_Wq
      launch_gemm(stream, rt, em_Wq, qb, kSEQ, 64, kD, kD, 64, 64, 0, 1, 1.f,
                  1, 16, (long long)kSEQ * kD, 1, 4, (long long)kD * 64,
                  1, 16, (long long)kSEQ * 64, 16);
    }
  }

  // 6) H += pm_read + sum_k read_acc
  combine_kernel<<<kTOK, 256, 0, stream>>>(Hb, pmr, ra);

  // 7) last 4 scan layers
  for (int l = 4; l < 8; l++)
    scan_layer(stream, Hb, xn, ub, gb,
               ln_g + (size_t)l * kD, ln_b + (size_t)l * kD,
               W_in + (size_t)l * kD * kDI, W_g + (size_t)l * kD * kDI,
               A_log + (size_t)l * kDI, W_out + (size_t)l * kDI * kD);

  // 8) final LN + tied-head logits: out @ embed^T * D^-0.5
  ln_rows<<<kTOK, 256, 0, stream>>>(Hb, lnf_g, lnf_b, xn);
  launch_gemm(stream, xn, embed, logits, kTOK, kV, kD, kD, kD, kV, 1, 0, 0.03125f,
              1, 1, 0, 1, 1, 0, 1, 1, 0, 1);
}
